// MultiHeadSelfAttention_46617575031442
// MI455X (gfx1250) — compile-verified, matrix operand order checked by reading, not run on hardware
//
#include <hip/hip_runtime.h>

typedef __attribute__((ext_vector_type(4)))  unsigned short u16x4;
typedef __attribute__((ext_vector_type(8)))  unsigned short u16x8;
typedef __attribute__((ext_vector_type(16))) unsigned short u16x16;
typedef __attribute__((ext_vector_type(16))) __bf16        v16bf;
typedef __attribute__((ext_vector_type(8)))  float         v8f;
typedef int i32x4 __attribute__((vector_size(16)));   // matches builtin param type

#define D_MODEL   1024
#define NUM_HEADS 16
#define D_K       64
#define SEQ       2048
#define BATCH     4

#if __has_builtin(__builtin_amdgcn_global_load_async_to_lds_b128) && \
    __has_builtin(__builtin_amdgcn_s_wait_asynccnt)
#define HAVE_ASYNC_LDS 1
#else
#define HAVE_ASYNC_LDS 0
#endif

#define AS_GLOBAL __attribute__((address_space(1)))
#define AS_LDS    __attribute__((address_space(3)))

__device__ __forceinline__ unsigned short f2bf(float x) {
  unsigned u = __builtin_bit_cast(unsigned, x);
  unsigned r = (u + 0x7FFFu + ((u >> 16) & 1u)) >> 16;
  return (unsigned short)r;
}
__device__ __forceinline__ float bf2f(unsigned short h) {
  unsigned u = ((unsigned)h) << 16;
  return __builtin_bit_cast(float, u);
}
__device__ __forceinline__ v16bf mkfrag(u16x8 lo, u16x8 hi) {
  u16x16 w = __builtin_shufflevector(lo, hi, 0,1,2,3,4,5,6,7,8,9,10,11,12,13,14,15);
  return __builtin_bit_cast(v16bf, w);
}

union AccF { v8f v; float f[8]; };

// ---------------- fp32 -> bf16 convert ----------------
__global__ void cvt_f32_bf16(const float* __restrict__ src,
                             unsigned short* __restrict__ dst, int n) {
  int i = blockIdx.x * 256 + threadIdx.x;
  if (i < n) dst[i] = f2bf(src[i]);
}

// ---------------- GEMM: C[M,N] = A[M,K](bf16) @ W[N,K]^T(bf16) ----------------
// block = 256 threads (8 waves), tile 128x128, BK=32, wave tile 64x32 (4x2 wmma)
template <typename OutT>
__global__ __launch_bounds__(256) void gemm_bf16_tn(
    const unsigned short* __restrict__ A,
    const unsigned short* __restrict__ W,
    OutT* __restrict__ C, int M, int N, int K) {
  __shared__ alignas(16) unsigned short As[128][40];
  __shared__ alignas(16) unsigned short Bs[128][40];
  const int tid  = threadIdx.x;
  const int lane = tid & 31;
  const int w    = tid >> 5;
  const int l16  = lane & 15;
  const int half = (lane >> 4) & 1;
  const int wm   = w >> 2;   // 0..1
  const int wn   = w & 3;    // 0..3
  const int m0   = blockIdx.y * 128;
  const int n0   = blockIdx.x * 128;

  AccF acc[4][2];
#pragma unroll
  for (int mt = 0; mt < 4; ++mt)
#pragma unroll
    for (int nt = 0; nt < 2; ++nt)
#pragma unroll
      for (int r = 0; r < 8; ++r) acc[mt][nt].f[r] = 0.0f;

  for (int k0 = 0; k0 < K; k0 += 32) {
    // stage tiles into LDS
#pragma unroll
    for (int i = 0; i < 2; ++i) {
      int c   = tid * 2 + i;          // 0..511
      int row = c >> 2;               // 0..127
      int col = (c & 3) * 8;          // 0,8,16,24
      const unsigned short* ga = &A[(size_t)(m0 + row) * K + k0 + col];
      const unsigned short* gw = &W[(size_t)(n0 + row) * K + k0 + col];
#if HAVE_ASYNC_LDS
      __builtin_amdgcn_global_load_async_to_lds_b128(
          (AS_GLOBAL i32x4*)(void*)ga, (AS_LDS i32x4*)&As[row][col], 0, 0);
      __builtin_amdgcn_global_load_async_to_lds_b128(
          (AS_GLOBAL i32x4*)(void*)gw, (AS_LDS i32x4*)&Bs[row][col], 0, 0);
#else
      *(u16x8*)&As[row][col] = *(const u16x8*)ga;
      *(u16x8*)&Bs[row][col] = *(const u16x8*)gw;
#endif
    }
#if HAVE_ASYNC_LDS
    __builtin_amdgcn_s_wait_asynccnt(0);
#endif
    __syncthreads();

    v16bf af[4], bfv[2];
#pragma unroll
    for (int mt = 0; mt < 4; ++mt) {
      int row = wm * 64 + mt * 16 + l16;
      u16x8 lo = *(const u16x8*)&As[row][half * 8];
      u16x8 hi = *(const u16x8*)&As[row][16 + half * 8];
      af[mt] = mkfrag(lo, hi);
    }
#pragma unroll
    for (int nt = 0; nt < 2; ++nt) {
      int row = wn * 32 + nt * 16 + l16;
      u16x8 lo = *(const u16x8*)&Bs[row][half * 8];
      u16x8 hi = *(const u16x8*)&Bs[row][16 + half * 8];
      bfv[nt] = mkfrag(lo, hi);
    }
#pragma unroll
    for (int mt = 0; mt < 4; ++mt)
#pragma unroll
      for (int nt = 0; nt < 2; ++nt)
        acc[mt][nt].v = __builtin_amdgcn_wmma_f32_16x16x32_bf16(
            false, af[mt], false, bfv[nt], (short)0, acc[mt][nt].v, false, false);
    __syncthreads();
  }

#pragma unroll
  for (int mt = 0; mt < 4; ++mt)
#pragma unroll
    for (int nt = 0; nt < 2; ++nt)
#pragma unroll
      for (int r = 0; r < 8; ++r) {
        int m = m0 + wm * 64 + mt * 16 + r + half * 8;
        int n = n0 + wn * 32 + nt * 16 + l16;
        float v = acc[mt][nt].f[r];
        if constexpr (sizeof(OutT) == 2) C[(size_t)m * N + n] = (OutT)f2bf(v);
        else                             C[(size_t)m * N + n] = (OutT)v;
      }
}

// ---------------- RoPE (in place on bf16 [B,S,H,DK]) ----------------
__global__ void rope_kernel(unsigned short* __restrict__ buf,
                            const int* __restrict__ pos) {
  int idx  = blockIdx.x * 256 + threadIdx.x; // B*S*H*32 threads, one rot pair each
  int pair = idx & 31;
  int h    = (idx >> 5) & 15;
  int s    = (idx >> 9) & 2047;
  int b    = idx >> 20;
  float p    = (float)pos[s];
  float freq = __expf(-((float)(2 * pair) * (1.0f / 64.0f)) * 9.2103403719761836f);
  float sn, c;
  __sincosf(p * freq, &sn, &c);
  size_t off = (((size_t)b * SEQ + s) * NUM_HEADS + h) * (size_t)D_K + 2 * pair;
  float x1 = bf2f(buf[off]);
  float x2 = bf2f(buf[off + 1]);
  buf[off]     = f2bf(x1 * c - x2 * sn);
  buf[off + 1] = f2bf(x1 * sn + x2 * c);
}

// ---------------- Flash attention (causal), 4 waves / 64-query block ----------------
__global__ __launch_bounds__(128) void attn_kernel(
    const unsigned short* __restrict__ Qp,
    const unsigned short* __restrict__ Kp,
    const unsigned short* __restrict__ Vp,
    unsigned short* __restrict__ ctx) {
  __shared__ alignas(16) unsigned short Vt[64][72];      // V^T tile: [d][kk]
  __shared__ alignas(16) unsigned short Ps[4][16][64];   // per-wave P staging
  const int tid  = threadIdx.x;
  const int lane = tid & 31;
  const int w    = tid >> 5;       // 0..3
  const int l16  = lane & 15;
  const int half = (lane >> 4) & 1;
  const int qi   = blockIdx.x;     // query tile (64 rows)
  const int bh   = blockIdx.y;
  const int b    = bh >> 4;
  const int h    = bh & 15;
  const int q0   = qi * 64 + w * 16;

  // Q fragments (held for whole kernel): A layout, lane = query row
  v16bf qf[2];
  {
    const size_t qbase = ((size_t)b * SEQ + q0 + l16) * D_MODEL + h * D_K;
#pragma unroll
    for (int c = 0; c < 2; ++c) {
      u16x8 lo = *(const u16x8*)&Qp[qbase + c * 32 + half * 8];
      u16x8 hi = *(const u16x8*)&Qp[qbase + c * 32 + 16 + half * 8];
      qf[c] = mkfrag(lo, hi);
    }
  }

  AccF o[4];
#pragma unroll
  for (int dt = 0; dt < 4; ++dt)
#pragma unroll
    for (int r = 0; r < 8; ++r) o[dt].f[r] = 0.0f;
  float mrow[8], lrow[8];
#pragma unroll
  for (int r = 0; r < 8; ++r) { mrow[r] = -__builtin_inff(); lrow[r] = 0.0f; }

  for (int kt = 0; kt <= qi; ++kt) {
    const int kbase = kt * 64;
    __syncthreads();
    {  // stage V tile transposed into LDS: 4 rows x 8 cols per thread,
       // packed ds_store_b64 (4 transposed elements per store)
      int grp  = tid >> 3;         // 0..15 -> rows grp*4 .. grp*4+3
      int seg  = (tid & 7) * 8;    // d segment
      int row0 = grp * 4;
      const size_t vb = ((size_t)b * SEQ + kbase) * D_MODEL + h * D_K + seg;
      u16x8 r0 = *(const u16x8*)&Vp[vb + (size_t)(row0 + 0) * D_MODEL];
      u16x8 r1 = *(const u16x8*)&Vp[vb + (size_t)(row0 + 1) * D_MODEL];
      u16x8 r2 = *(const u16x8*)&Vp[vb + (size_t)(row0 + 2) * D_MODEL];
      u16x8 r3 = *(const u16x8*)&Vp[vb + (size_t)(row0 + 3) * D_MODEL];
#pragma unroll
      for (int e = 0; e < 8; ++e) {
        u16x4 p;
        p.x = r0[e]; p.y = r1[e]; p.z = r2[e]; p.w = r3[e];
        *(u16x4*)&Vt[seg + e][row0] = p;
      }
    }
    // prefetch next key tile (K and V rows) into cache
    if (kt + 1 <= qi) {
      const size_t nb = ((size_t)b * SEQ + kbase + 64 + (tid >> 1)) * D_MODEL + h * D_K;
      __builtin_prefetch(&Kp[nb], 0, 1);
      __builtin_prefetch(&Vp[nb], 0, 1);
    }
    __syncthreads();

    // ---- scores: S = Q @ K^T (contraction over d, K frags direct from global)
    AccF s[4];
#pragma unroll
    for (int t = 0; t < 4; ++t)
#pragma unroll
      for (int r = 0; r < 8; ++r) s[t].f[r] = 0.0f;
#pragma unroll
    for (int t = 0; t < 4; ++t) {
      const size_t kb = ((size_t)b * SEQ + kbase + t * 16 + l16) * D_MODEL + h * D_K;
#pragma unroll
      for (int c = 0; c < 2; ++c) {
        u16x8 lo = *(const u16x8*)&Kp[kb + c * 32 + half * 8];
        u16x8 hi = *(const u16x8*)&Kp[kb + c * 32 + 16 + half * 8];
        v16bf kf = mkfrag(lo, hi);
        s[t].v = __builtin_amdgcn_wmma_f32_16x16x32_bf16(
            false, qf[c], false, kf, (short)0, s[t].v, false, false);
      }
    }

    // ---- scale + causal mask (diagonal tile only)
    const bool diag = (kt == qi);
#pragma unroll
    for (int t = 0; t < 4; ++t)
#pragma unroll
      for (int r = 0; r < 8; ++r) {
        float v = s[t].f[r] * 0.125f;
        if (diag && (kbase + t * 16 + l16) > (q0 + r + half * 8)) v = -__builtin_inff();
        s[t].f[r] = v;
      }

    // ---- online softmax (row = r + 8*half; 16-lane groups hold one row)
    float factor[8];
#pragma unroll
    for (int r = 0; r < 8; ++r) {
      float mx = fmaxf(fmaxf(s[0].f[r], s[1].f[r]), fmaxf(s[2].f[r], s[3].f[r]));
#pragma unroll
      for (int msk = 1; msk < 16; msk <<= 1) mx = fmaxf(mx, __shfl_xor(mx, msk, 32));
      float mnew = fmaxf(mrow[r], mx);
      float corr = __expf(mrow[r] - mnew);
      float rs = 0.0f;
#pragma unroll
      for (int t = 0; t < 4; ++t) {
        float pe = __expf(s[t].f[r] - mnew);
        s[t].f[r] = pe;
        rs += pe;
      }
#pragma unroll
      for (int msk = 1; msk < 16; msk <<= 1) rs += __shfl_xor(rs, msk, 32);
      lrow[r]   = lrow[r] * corr + rs;
      mrow[r]   = mnew;
      factor[r] = corr;
    }
#pragma unroll
    for (int dt = 0; dt < 4; ++dt)
#pragma unroll
      for (int r = 0; r < 8; ++r) o[dt].f[r] *= factor[r];

    // ---- stage P through wave-private LDS (C-layout -> A-layout)
#pragma unroll
    for (int t = 0; t < 4; ++t)
#pragma unroll
      for (int r = 0; r < 8; ++r)
        Ps[w][r + half * 8][t * 16 + l16] = f2bf(s[t].f[r]);

    v16bf pf[2];
#pragma unroll
    for (int c = 0; c < 2; ++c) {
      u16x8 lo = *(const u16x8*)&Ps[w][l16][c * 32 + half * 8];
      u16x8 hi = *(const u16x8*)&Ps[w][l16][c * 32 + 16 + half * 8];
      pf[c] = mkfrag(lo, hi);
    }

    // ---- O += P @ V (contraction over kk; V^T tile gives contiguous kk reads)
#pragma unroll
    for (int dt = 0; dt < 4; ++dt)
#pragma unroll
      for (int c = 0; c < 2; ++c) {
        u16x8 lo = *(const u16x8*)&Vt[dt * 16 + l16][c * 32 + half * 8];
        u16x8 hi = *(const u16x8*)&Vt[dt * 16 + l16][c * 32 + 16 + half * 8];
        v16bf vf = mkfrag(lo, hi);
        o[dt].v = __builtin_amdgcn_wmma_f32_16x16x32_bf16(
            false, pf[c], false, vf, (short)0, o[dt].v, false, false);
      }
  }

  // ---- normalize and write ctx (bf16, [b,s,h*64+d])
#pragma unroll
  for (int r = 0; r < 8; ++r) {
    float inv = 1.0f / lrow[r];
    size_t base = ((size_t)b * SEQ + q0 + r + half * 8) * D_MODEL + h * D_K;
#pragma unroll
    for (int dt = 0; dt < 4; ++dt)
      ctx[base + dt * 16 + l16] = f2bf(o[dt].f[r] * inv);
  }
}

// ---------------- launcher ----------------
extern "C" void kernel_launch(void* const* d_in, const int* in_sizes, int n_in,
                              void* d_out, int out_size, void* d_ws, size_t ws_size,
                              hipStream_t stream) {
  const float* x  = (const float*)d_in[0];
  const float* Wq = (const float*)d_in[1];
  const float* Wk = (const float*)d_in[2];
  const float* Wv = (const float*)d_in[3];
  const float* Wo = (const float*)d_in[4];
  const int*  tok = (const int*)d_in[5];
  float* out = (float*)d_out;

  const size_t XB = (size_t)BATCH * SEQ * D_MODEL;  // 8388608
  const size_t WB = (size_t)D_MODEL * D_MODEL;      // 1048576

  char* ws = (char*)d_ws;
  size_t off = 0;
  unsigned short* xb  = (unsigned short*)(ws + off); off += XB * 2;
  unsigned short* wqb = (unsigned short*)(ws + off); off += WB * 2;
  unsigned short* wkb = (unsigned short*)(ws + off); off += WB * 2;
  unsigned short* wvb = (unsigned short*)(ws + off); off += WB * 2;
  unsigned short* wob = (unsigned short*)(ws + off); off += WB * 2;
  unsigned short* Qp  = (unsigned short*)(ws + off); off += XB * 2;
  unsigned short* Kp  = (unsigned short*)(ws + off); off += XB * 2;
  unsigned short* Vp  = (unsigned short*)(ws + off); off += XB * 2;
  unsigned short* ctx = (unsigned short*)(ws + off); off += XB * 2;
  if (ws_size < off) return;  // need ~88 MB of scratch

  cvt_f32_bf16<<<(int)((XB + 255) / 256), 256, 0, stream>>>(x, xb, (int)XB);
  cvt_f32_bf16<<<(int)((WB + 255) / 256), 256, 0, stream>>>(Wq, wqb, (int)WB);
  cvt_f32_bf16<<<(int)((WB + 255) / 256), 256, 0, stream>>>(Wk, wkb, (int)WB);
  cvt_f32_bf16<<<(int)((WB + 255) / 256), 256, 0, stream>>>(Wv, wvb, (int)WB);
  cvt_f32_bf16<<<(int)((WB + 255) / 256), 256, 0, stream>>>(Wo, wob, (int)WB);

  const int M = BATCH * SEQ;  // 8192
  dim3 ggrid(D_MODEL / 128, M / 128);  // (8, 64)
  gemm_bf16_tn<unsigned short><<<ggrid, 256, 0, stream>>>(xb, wqb, Qp, M, D_MODEL, D_MODEL);
  gemm_bf16_tn<unsigned short><<<ggrid, 256, 0, stream>>>(xb, wkb, Kp, M, D_MODEL, D_MODEL);
  gemm_bf16_tn<unsigned short><<<ggrid, 256, 0, stream>>>(xb, wvb, Vp, M, D_MODEL, D_MODEL);

  const int rope_total = BATCH * SEQ * NUM_HEADS * 32;  // 4194304
  rope_kernel<<<rope_total / 256, 256, 0, stream>>>(Qp, tok);
  rope_kernel<<<rope_total / 256, 256, 0, stream>>>(Kp, tok);

  dim3 agrid(SEQ / 64, BATCH * NUM_HEADS);  // (32, 64)
  attn_kernel<<<agrid, 128, 0, stream>>>(Qp, Kp, Vp, ctx);

  gemm_bf16_tn<float><<<ggrid, 256, 0, stream>>>(ctx, wob, out, M, D_MODEL, D_MODEL);
}
